// LatentAttentionHead_62869731279276
// MI455X (gfx1250) — compile-verified, matrix operand order checked by reading, not run on hardware
//
#include <hip/hip_runtime.h>
#include <hip/hip_bf16.h>

typedef _Float16 v16h __attribute__((ext_vector_type(16)));
typedef float    v8f  __attribute__((ext_vector_type(8)));
typedef float    v4f  __attribute__((ext_vector_type(4)));

#define WMMA_F32_F16(A, B, C) \
  __builtin_amdgcn_wmma_f32_16x16x32_f16(false, (A), false, (B), (short)0, (C), false, false)

static constexpr int   Bn    = 8;
static constexpr int   S     = 8192;
static constexpr int   E     = 1024;
static constexpr int   L     = 64;
static constexpr int   H     = 64;
static constexpr int   MROWS = Bn * S;      // 65536 total (b,s) rows
static constexpr float SCALE = 0.125f;      // 1/sqrt(64)

// ---------------------------------------------------------------------------
// Kernel 1: convert Wk, Wv (64x1024 f32) to f16 once; they then live in L2.
// ---------------------------------------------------------------------------
__global__ void cvt_weights(const float* __restrict__ Wk, const float* __restrict__ Wv,
                            _Float16* __restrict__ wk16, _Float16* __restrict__ wv16) {
  int i = blockIdx.x * blockDim.x + threadIdx.x;   // exactly 65536 threads
  wk16[i] = (_Float16)Wk[i];
  wv16[i] = (_Float16)Wv[i];
}

// ---------------------------------------------------------------------------
// Kernel 2: q[l,h] = scale * sum_d latents[l,d] * Wq[h,d]   (tiny: 64x64x256)
// stored f16 row-major [L][H], pre-scaled by 1/sqrt(H).
// ---------------------------------------------------------------------------
__global__ void compute_q(const float* __restrict__ latents, const float* __restrict__ Wq,
                          _Float16* __restrict__ q16) {
  int t = threadIdx.x;
  for (int o = t; o < L * H; o += 256) {
    int l = o >> 6, h = o & 63;
    float s = 0.f;
    for (int d = 0; d < 256; ++d) s += latents[l * 256 + d] * Wq[h * 256 + d];
    q16[o] = (_Float16)(s * SCALE);
  }
}

// ---------------------------------------------------------------------------
// Kernel 3: K/V projection. GEMM M=65536, N=128(K|V), Kred=1024.
// 512 WGs x 8 waves; each wave: 16 rows x 128 cols, K-loop in steps of 32.
// Software pipelined: all 8 weight-fragment loads for step e issue as one
// clause, the raw input block for step e+32 is in flight during the WMMAs.
// Input stream is non-temporal (read once; keep L2 for weights + K/V).
// ---------------------------------------------------------------------------
__global__ __launch_bounds__(256) void kv_proj(const float* __restrict__ x,
        const _Float16* __restrict__ wk16, const _Float16* __restrict__ wv16,
        _Float16* __restrict__ kws, _Float16* __restrict__ vt) {
  const int tid  = threadIdx.x;
  const int w    = tid >> 5, lane = tid & 31;
  const int lg   = lane >> 4, ln  = lane & 15;
  const size_t rowBase = (size_t)blockIdx.x * 128 + (size_t)w * 16;
  const float* arow = x + (rowBase + ln) * E;
  const int col0 = lg * 16;

  v8f ak[4] = {};
  v8f av[4] = {};

  // preload first raw A block (16 f32 per lane)
  v4f f0 = __builtin_nontemporal_load((const v4f*)(arow + col0));
  v4f f1 = __builtin_nontemporal_load((const v4f*)(arow + col0 + 4));
  v4f f2 = __builtin_nontemporal_load((const v4f*)(arow + col0 + 8));
  v4f f3 = __builtin_nontemporal_load((const v4f*)(arow + col0 + 12));

  for (int e = 0; e < E; e += 32) {
    const int eo = e + col0;
    // all B fragments for this step in one clause (cache-resident weights)
    v16h bk[4], bv[4];
    #pragma unroll
    for (int nt = 0; nt < 4; ++nt)
      bk[nt] = *(const v16h*)(wk16 + (nt * 16 + ln) * E + eo);
    #pragma unroll
    for (int nt = 0; nt < 4; ++nt)
      bv[nt] = *(const v16h*)(wv16 + (nt * 16 + ln) * E + eo);

    // next-step raw A block (masked address: last iteration re-reads col 0,
    // stays in-bounds, result discarded)
    const int en = ((e + 32) & (E - 1)) + col0;
    v4f g0 = __builtin_nontemporal_load((const v4f*)(arow + en));
    v4f g1 = __builtin_nontemporal_load((const v4f*)(arow + en + 4));
    v4f g2 = __builtin_nontemporal_load((const v4f*)(arow + en + 8));
    v4f g3 = __builtin_nontemporal_load((const v4f*)(arow + en + 12));

    // convert current A block to f16
    v16h a;
    #pragma unroll
    for (int i = 0; i < 4; ++i) {
      a[i]      = (_Float16)f0[i];
      a[4 + i]  = (_Float16)f1[i];
      a[8 + i]  = (_Float16)f2[i];
      a[12 + i] = (_Float16)f3[i];
    }

    #pragma unroll
    for (int nt = 0; nt < 4; ++nt) ak[nt] = WMMA_F32_F16(a, bk[nt], ak[nt]);
    #pragma unroll
    for (int nt = 0; nt < 4; ++nt) av[nt] = WMMA_F32_F16(a, bv[nt], av[nt]);

    f0 = g0; f1 = g1; f2 = g2; f3 = g3;
  }

  // k: row-major [s][64]; element (M = r+8*lg, N = nt*16+ln)
  #pragma unroll
  for (int nt = 0; nt < 4; ++nt)
    #pragma unroll
    for (int r = 0; r < 8; ++r)
      kws[(rowBase + r + 8 * lg) * 64 + nt * 16 + ln] = (_Float16)ak[nt][r];

  // v: transposed [h][MROWS]; per lane 8 consecutive s -> one 16B store
  #pragma unroll
  for (int nt = 0; nt < 4; ++nt) {
    _Float16 tmp[8] __attribute__((aligned(16)));
    #pragma unroll
    for (int r = 0; r < 8; ++r) tmp[r] = (_Float16)av[nt][r];
    const int h = nt * 16 + ln;
    *(v4f*)(vt + (size_t)h * MROWS + rowBase + 8 * lg) = *(const v4f*)tmp;
  }
}

// ---------------------------------------------------------------------------
// Kernel 4: attention partials (flash-style). Grid = 8 batches x 16 chunks.
// Each of 8 waves handles a 64-wide s-slice: scores (WMMA), per-row softmax
// stats via 16-lane shfl_xor reductions, P repacked via LDS into A-fragments,
// P@V (WMMA). Waves then merge (m, l, o) in LDS and write one partial per WG.
// K/V fragment reads are non-temporal (each byte consumed exactly once).
// ---------------------------------------------------------------------------
__global__ __launch_bounds__(256) void attn_partial(const _Float16* __restrict__ q16,
        const _Float16* __restrict__ kws, const _Float16* __restrict__ vt,
        float* __restrict__ pm, float* __restrict__ plsum, float* __restrict__ po) {
  __shared__ unsigned int smem[16384];   // 64 KB, reused across phases
  const int tid   = threadIdx.x;
  const int w     = tid >> 5, lane = tid & 31;
  const int lg    = lane >> 4, ln  = lane & 15;
  const int b     = blockIdx.x >> 4;     // 16 chunks per batch
  const int chunk = blockIdx.x & 15;
  const int s0    = chunk * 512 + w * 64;  // within batch

  // q fragments (A-matrix, M=l): [mtile][kstep]
  v16h qa[4][2];
  #pragma unroll
  for (int mt = 0; mt < 4; ++mt)
    #pragma unroll
    for (int kk = 0; kk < 2; ++kk)
      qa[mt][kk] = *(const v16h*)(q16 + (mt * 16 + ln) * 64 + kk * 32 + lg * 16);

  // scores = q @ k^T over this wave's 64 s-columns
  v8f acc[4][4] = {};
  #pragma unroll
  for (int nt = 0; nt < 4; ++nt) {
    const _Float16* kr = kws + ((size_t)b * S + s0 + nt * 16 + ln) * 64 + lg * 16;
    v16h kb0 = __builtin_nontemporal_load((const v16h*)kr);
    v16h kb1 = __builtin_nontemporal_load((const v16h*)(kr + 32));
    #pragma unroll
    for (int mt = 0; mt < 4; ++mt) acc[mt][nt] = WMMA_F32_F16(qa[mt][0], kb0, acc[mt][nt]);
    #pragma unroll
    for (int mt = 0; mt < 4; ++mt) acc[mt][nt] = WMMA_F32_F16(qa[mt][1], kb1, acc[mt][nt]);
  }

  // per-row (l) max & sum within the slice; overwrite acc with p = exp(s - m)
  float rmax[4][8], rsum[4][8];
  _Float16* p16 = (_Float16*)smem + w * 4096;   // wave-private 8 KB
  #pragma unroll
  for (int mt = 0; mt < 4; ++mt)
    #pragma unroll
    for (int r = 0; r < 8; ++r) {
      float mx = acc[mt][0][r];
      #pragma unroll
      for (int nt = 1; nt < 4; ++nt) mx = fmaxf(mx, acc[mt][nt][r]);
      #pragma unroll
      for (int off = 1; off < 16; off <<= 1) mx = fmaxf(mx, __shfl_xor(mx, off, 32));
      float sum = 0.f;
      #pragma unroll
      for (int nt = 0; nt < 4; ++nt) {
        float p = __expf(acc[mt][nt][r] - mx);
        acc[mt][nt][r] = p;
        sum += p;
      }
      #pragma unroll
      for (int off = 1; off < 16; off <<= 1) sum += __shfl_xor(sum, off, 32);
      rmax[mt][r] = mx;
      rsum[mt][r] = sum;
      #pragma unroll
      for (int nt = 0; nt < 4; ++nt)   // repack p (row l, col s) into LDS
        p16[(mt * 16 + r + 8 * lg) * 64 + nt * 16 + ln] = (_Float16)acc[mt][nt][r];
    }

  // o = P @ V  (A from LDS, B contiguous from transposed v)
  v8f oacc[4][4] = {};
  #pragma unroll
  for (int kk = 0; kk < 2; ++kk) {
    v16h pa[4];
    #pragma unroll
    for (int mt = 0; mt < 4; ++mt)
      pa[mt] = *(const v16h*)(p16 + (mt * 16 + ln) * 64 + kk * 32 + lg * 16);
    v16h vbt[4];
    #pragma unroll
    for (int nt = 0; nt < 4; ++nt)
      vbt[nt] = __builtin_nontemporal_load(
          (const v16h*)(vt + (size_t)(nt * 16 + ln) * MROWS +
                        (size_t)b * S + s0 + kk * 32 + lg * 16));
    #pragma unroll
    for (int nt = 0; nt < 4; ++nt)
      #pragma unroll
      for (int mt = 0; mt < 4; ++mt)
        oacc[mt][nt] = WMMA_F32_F16(pa[mt], vbt[nt], oacc[mt][nt]);
  }

  // --- cross-wave merge in LDS (p region is now dead) ---
  __syncthreads();
  float* sf = (float*)smem;
  // [0,4096): o  [4096,4160): lsum  [4160,4672): per-wave m  [4672,4736): Mwg
  for (int i = tid; i < 4096; i += 256) sf[i] = 0.f;
  if (tid < 64) sf[4096 + tid] = 0.f;
  if (ln == 0) {
    #pragma unroll
    for (int mt = 0; mt < 4; ++mt)
      #pragma unroll
      for (int r = 0; r < 8; ++r)
        sf[4160 + w * 64 + mt * 16 + r + 8 * lg] = rmax[mt][r];
  }
  __syncthreads();
  if (tid < 64) {
    float M = -3.4e38f;
    #pragma unroll
    for (int ww = 0; ww < 8; ++ww) M = fmaxf(M, sf[4160 + ww * 64 + tid]);
    sf[4672 + tid] = M;
  }
  __syncthreads();
  #pragma unroll
  for (int mt = 0; mt < 4; ++mt)
    #pragma unroll
    for (int r = 0; r < 8; ++r) {
      const int   l = mt * 16 + r + 8 * lg;
      const float f = __expf(rmax[mt][r] - sf[4672 + l]);
      if (ln == 0) atomicAdd(&sf[4096 + l], rsum[mt][r] * f);
      #pragma unroll
      for (int nt = 0; nt < 4; ++nt)
        atomicAdd(&sf[l * 64 + nt * 16 + ln], oacc[mt][nt][r] * f);
    }
  __syncthreads();
  const size_t wg = blockIdx.x;
  for (int i = tid; i < 4096; i += 256) po[wg * 4096 + i] = sf[i];
  if (tid < 64) {
    pm[wg * 64 + tid]    = sf[4672 + tid];
    plsum[wg * 64 + tid] = sf[4096 + tid];
  }
}

// ---------------------------------------------------------------------------
// Kernel 5: merge 16 chunk partials per batch with max-rescaled softmax.
// ---------------------------------------------------------------------------
__global__ void attn_combine(const float* __restrict__ pm, const float* __restrict__ plsum,
                             const float* __restrict__ po, float* __restrict__ out) {
  __shared__ float Ms[64], Ls[64];
  const int b = blockIdx.x, t = threadIdx.x;
  if (t < 64) {
    float M = -3.4e38f;
    for (int c = 0; c < 16; ++c) M = fmaxf(M, pm[(b * 16 + c) * 64 + t]);
    float Lsum = 0.f;
    for (int c = 0; c < 16; ++c)
      Lsum += __expf(pm[(b * 16 + c) * 64 + t] - M) * plsum[(b * 16 + c) * 64 + t];
    Ms[t] = M;
    Ls[t] = Lsum;
  }
  __syncthreads();
  for (int i = t; i < 4096; i += 256) {
    const int   l = i >> 6;
    const float M = Ms[l];
    float acc = 0.f;
    for (int c = 0; c < 16; ++c)
      acc += __expf(pm[(b * 16 + c) * 64 + l] - M) * po[(size_t)(b * 16 + c) * 4096 + i];
    out[(size_t)b * 4096 + i] = acc / Ls[l];
  }
}

// ---------------------------------------------------------------------------
extern "C" void kernel_launch(void* const* d_in, const int* in_sizes, int n_in,
                              void* d_out, int out_size, void* d_ws, size_t ws_size,
                              hipStream_t stream) {
  (void)in_sizes; (void)n_in; (void)out_size; (void)ws_size;
  const float* x   = (const float*)d_in[0];   // [8,8192,1024]
  const float* lat = (const float*)d_in[1];   // [64,256]
  const float* Wq  = (const float*)d_in[2];   // [64,256]
  const float* Wk  = (const float*)d_in[3];   // [64,1024]
  const float* Wv  = (const float*)d_in[4];   // [64,1024]
  float* out = (float*)d_out;                 // [8,64,64]

  char* ws = (char*)d_ws;
  _Float16* q16  = (_Float16*)(ws + 0);                    //   8 KB
  _Float16* wk16 = (_Float16*)(ws + 8192);                 // 128 KB
  _Float16* wv16 = (_Float16*)(ws + 139264);               // 128 KB
  _Float16* kws  = (_Float16*)(ws + 270336);               //   8 MB  [s][64]
  _Float16* vtp  = (_Float16*)(ws + 8658944);              //   8 MB  [h][65536]
  float*    pm   = (float*)(ws + 17047552);                //  32 KB
  float*    pls  = (float*)(ws + 17080320);                //  32 KB
  float*    po   = (float*)(ws + 17113088);                //   2 MB
  // total ~18.3 MB of workspace

  cvt_weights <<<256, 256, 0, stream>>>(Wk, Wv, wk16, wv16);
  compute_q   <<<1,   256, 0, stream>>>(lat, Wq, q16);
  kv_proj     <<<512, 256, 0, stream>>>(x, wk16, wv16, kws, vtp);
  attn_partial<<<128, 256, 0, stream>>>(q16, kws, vtp, pm, pls, po);
  attn_combine<<<8,   256, 0, stream>>>(pm, pls, po, out);
}
